// TransformerBlock_49374944035441
// MI455X (gfx1250) — matrix operand order repaired and checked by reading, not compiled, not run on hardware
//
#include <hip/hip_runtime.h>

// ---------------------------------------------------------------------------
// Problem constants (match the reference)
// ---------------------------------------------------------------------------
#define DM   1024     // d_model
#define TSEQ 2048     // sequence length
#define BATCH 4
#define QHEADS 16
#define KVHEADS 4
#define HDIM 64
#define DFF  2048     // 2*DM

typedef __attribute__((ext_vector_type(16))) __bf16 v16bf;
typedef __attribute__((ext_vector_type(8)))  __bf16 v8bf;
typedef __attribute__((ext_vector_type(8)))  float  v8f;

// ---------------------------------------------------------------------------
// WMMA helper (CDNA5 16x16x32 bf16, f32 accumulate)
// ---------------------------------------------------------------------------
__device__ __forceinline__ v8f wmma_bf16(v16bf a, v16bf b, v8f c) {
  // (neg_a, A, neg_b, B, c_mod, C, reuse_a, reuse_b)
  return __builtin_amdgcn_wmma_f32_16x16x32_bf16(false, a, false, b, (short)0, c,
                                                 false, false);
}

// A-matrix fragment (16x32, M x K), row-major source with leading dim ld.
// Lane l<16:  row M=l,    K = {0..15}
// Lane l>=16: row M=l-16, K = {16..31}
__device__ __forceinline__ v16bf load_a_frag(const __bf16* __restrict__ base,
                                             int ld, int m0, int k0, int lane) {
  const __bf16* p = base + (size_t)(m0 + (lane & 15)) * ld + k0 + ((lane >> 4) << 4);
  v8bf lo = *reinterpret_cast<const v8bf*>(p);        // K offs +0..7
  v8bf hi = *reinterpret_cast<const v8bf*>(p + 8);    // K offs +8..15
  return __builtin_shufflevector(lo, hi, 0,1,2,3,4,5,6,7,8,9,10,11,12,13,14,15);
}

// B-matrix fragment (32x16, K x N) where memory holds W[N][K] row-major
// (i.e. B(k,n) = W[n][k]).  Per lane: column n = n0+(lane&15), 16 consecutive
// K values starting at k0 + (lane>=16 ? 16 : 0)  ->  contiguous 32 bytes.
__device__ __forceinline__ v16bf load_b_frag(const __bf16* __restrict__ base,
                                             int ld, int n0, int k0, int lane) {
  const __bf16* p = base + (size_t)(n0 + (lane & 15)) * ld + k0 + ((lane >> 4) << 4);
  v8bf lo = *reinterpret_cast<const v8bf*>(p);
  v8bf hi = *reinterpret_cast<const v8bf*>(p + 8);
  return __builtin_shufflevector(lo, hi, 0,1,2,3,4,5,6,7,8,9,10,11,12,13,14,15);
}

// ---------------------------------------------------------------------------
// fp32 -> bf16 elementwise conversion (weights)
// ---------------------------------------------------------------------------
__global__ void cvt_f32_bf16_kernel(const float* __restrict__ in,
                                    __bf16* __restrict__ out, int n) {
  int i = blockIdx.x * blockDim.x + threadIdx.x;
  if (i < n) out[i] = (__bf16)in[i];
}

// ---------------------------------------------------------------------------
// LayerNorm over last dim (C = 1024), one block (256 thr) per row, bf16 out
// ---------------------------------------------------------------------------
__global__ __launch_bounds__(256) void layernorm_kernel(
    const float* __restrict__ x, const float* __restrict__ g,
    const float* __restrict__ b, __bf16* __restrict__ out) {
  __shared__ float red[8];
  const int row = blockIdx.x;
  const float* xr = x + (size_t)row * DM;
  float vals[4];
  float lsum = 0.f;
  #pragma unroll
  for (int i = 0; i < 4; ++i) {
    vals[i] = xr[threadIdx.x + i * 256];
    lsum += vals[i];
  }
  #pragma unroll
  for (int off = 16; off > 0; off >>= 1) lsum += __shfl_xor(lsum, off, 32);
  if ((threadIdx.x & 31) == 0) red[threadIdx.x >> 5] = lsum;
  __syncthreads();
  float mean = 0.f;
  #pragma unroll
  for (int i = 0; i < 8; ++i) mean += red[i];
  mean *= (1.0f / DM);
  __syncthreads();
  float lvar = 0.f;
  #pragma unroll
  for (int i = 0; i < 4; ++i) { float d = vals[i] - mean; lvar += d * d; }
  #pragma unroll
  for (int off = 16; off > 0; off >>= 1) lvar += __shfl_xor(lvar, off, 32);
  if ((threadIdx.x & 31) == 0) red[threadIdx.x >> 5] = lvar;
  __syncthreads();
  float var = 0.f;
  #pragma unroll
  for (int i = 0; i < 8; ++i) var += red[i];
  var *= (1.0f / DM);
  const float rs = rsqrtf(var + 1e-5f);
  #pragma unroll
  for (int i = 0; i < 4; ++i) {
    int c = threadIdx.x + i * 256;
    out[(size_t)row * DM + c] = (__bf16)((vals[i] - mean) * rs * g[c] + b[c]);
  }
}

// ---------------------------------------------------------------------------
// Generic bf16 WMMA GEMM:  C[M,N] = A[M,K] * W[N,K]^T  (+ bias / resid / gelu)
// Block = 128 threads = 4 waves (2x2); block tile 128x128; wave tile 64x64
// = 4x4 WMMA tiles -> 16 v_wmma per 32-wide K step against 16 b128 loads.
// ---------------------------------------------------------------------------
enum { EPI_F32 = 0, EPI_RES_F32 = 1, EPI_GELU_BF16 = 2 };

template <int EPI>
__global__ __launch_bounds__(128) void gemm_bf16_kernel(
    const __bf16* __restrict__ A, const __bf16* __restrict__ W,
    const float* __restrict__ bias, const float* __restrict__ resid,
    float* __restrict__ outF, __bf16* __restrict__ outB,
    int M, int N, int K) {
  const int lane = threadIdx.x & 31;
  const int wave = threadIdx.x >> 5;
  const int wm = wave & 1;          // 2 waves along M
  const int wn = wave >> 1;         // 2 waves along N
  const int m0 = blockIdx.y * 128 + wm * 64;
  const int n0 = blockIdx.x * 128 + wn * 64;

  v8f acc[4][4] = {};
  for (int k0 = 0; k0 < K; k0 += 32) {
    v16bf a0 = load_a_frag(A, K, m0,      k0, lane);
    v16bf a1 = load_a_frag(A, K, m0 + 16, k0, lane);
    v16bf a2 = load_a_frag(A, K, m0 + 32, k0, lane);
    v16bf a3 = load_a_frag(A, K, m0 + 48, k0, lane);
    #pragma unroll
    for (int j = 0; j < 4; ++j) {
      v16bf b = load_b_frag(W, K, n0 + j * 16, k0, lane);
      acc[0][j] = wmma_bf16(a0, b, acc[0][j]);
      acc[1][j] = wmma_bf16(a1, b, acc[1][j]);
      acc[2][j] = wmma_bf16(a2, b, acc[2][j]);
      acc[3][j] = wmma_bf16(a3, b, acc[3][j]);
    }
    // L2 prefetch of the next K-panel (global_prefetch_b8)
    if (k0 + 32 < K) {
      __builtin_prefetch(A + (size_t)(m0 + (lane & 15)) * K + k0 + 32, 0, 3);
      __builtin_prefetch(W + (size_t)(n0 + (lane & 15)) * K + k0 + 32, 0, 3);
    }
  }

  const int nlo = lane & 15;
  const int half = lane >> 4;
  #pragma unroll
  for (int i = 0; i < 4; ++i) {
    #pragma unroll
    for (int j = 0; j < 4; ++j) {
      #pragma unroll
      for (int r = 0; r < 8; ++r) {
        const int m = m0 + i * 16 + half * 8 + r;
        const int n = n0 + j * 16 + nlo;
        float v = acc[i][j][r];
        if (EPI == EPI_F32) {
          outF[(size_t)m * N + n] = v;
        } else if (EPI == EPI_RES_F32) {
          const float bb = bias ? bias[n] : 0.0f;
          outF[(size_t)m * N + n] = resid[(size_t)m * N + n] + v + bb;
        } else {  // EPI_GELU_BF16 (exact gelu: 0.5*x*(1+erf(x/sqrt(2))))
          const float u = v + (bias ? bias[n] : 0.0f);
          const float ge = 0.5f * u * (1.0f + erff(u * 0.70710678118654752f));
          outB[(size_t)m * N + n] = (__bf16)ge;
        }
      }
    }
  }
}

// ---------------------------------------------------------------------------
// RoPE kernels + V transpose (fp32 projections -> bf16 head-major tensors)
// ---------------------------------------------------------------------------
__global__ void rope_q_kernel(const float* __restrict__ qlin,
                              const float* __restrict__ cosT,
                              const float* __restrict__ sinT,
                              __bf16* __restrict__ Qr) {
  int id = blockIdx.x * blockDim.x + threadIdx.x;       // B*QH*T*32
  if (id >= BATCH * QHEADS * TSEQ * (HDIM / 2)) return;
  const int p = id & 31;
  const int rest = id >> 5;
  const int t = rest & (TSEQ - 1);
  const int bh = rest >> 11;                            // b*QH + qh
  const int qh = bh & (QHEADS - 1);
  const int b = bh >> 4;
  const float* src = qlin + ((size_t)(b * TSEQ + t)) * DM + qh * HDIM + 2 * p;
  const float x0 = src[0], x1 = src[1];
  const float c = cosT[t * HDIM + 2 * p];
  const float s = sinT[t * HDIM + 2 * p];
  __bf16* dst = Qr + (((size_t)bh * TSEQ) + t) * HDIM + 2 * p;
  dst[0] = (__bf16)(x0 * c - x1 * s);
  dst[1] = (__bf16)(x1 * c + x0 * s);
}

__global__ void rope_k_kernel(const float* __restrict__ kvlin,
                              const float* __restrict__ cosT,
                              const float* __restrict__ sinT,
                              __bf16* __restrict__ Kr) {
  int id = blockIdx.x * blockDim.x + threadIdx.x;       // B*KVH*T*32
  if (id >= BATCH * KVHEADS * TSEQ * (HDIM / 2)) return;
  const int p = id & 31;
  const int rest = id >> 5;
  const int t = rest & (TSEQ - 1);
  const int bh = rest >> 11;                            // b*KVH + kh
  const int kh = bh & (KVHEADS - 1);
  const int b = bh >> 2;
  const float* src =
      kvlin + ((size_t)(b * TSEQ + t)) * (2 * KVHEADS * HDIM) + kh * 128 + 2 * p;
  const float x0 = src[0], x1 = src[1];
  const float c = cosT[t * HDIM + 2 * p];
  const float s = sinT[t * HDIM + 2 * p];
  __bf16* dst = Kr + (((size_t)bh * TSEQ) + t) * HDIM + 2 * p;
  dst[0] = (__bf16)(x0 * c - x1 * s);
  dst[1] = (__bf16)(x1 * c + x0 * s);
}

// Vt[b, kh, d, t]  <-  kv[b, t, kh*128 + 64 + d]   (d-major so P@V B-frags
// are contiguous 32B loads per lane)
__global__ void v_transpose_kernel(const float* __restrict__ kvlin,
                                   __bf16* __restrict__ Vt) {
  int id = blockIdx.x * blockDim.x + threadIdx.x;       // B*KVH*HD*T
  if (id >= BATCH * KVHEADS * HDIM * TSEQ) return;
  const int t = id & (TSEQ - 1);
  const int d = (id >> 11) & (HDIM - 1);
  const int bh = id >> 17;                              // b*KVH + kh
  const int kh = bh & (KVHEADS - 1);
  const int b = bh >> 2;
  const float v =
      kvlin[((size_t)(b * TSEQ + t)) * (2 * KVHEADS * HDIM) + kh * 128 + 64 + d];
  Vt[(((size_t)bh * HDIM) + d) * TSEQ + t] = (__bf16)v;
}

// ---------------------------------------------------------------------------
// Attention: one wave per 16-query tile.  Two passes through the key axis:
//   pass1: S = (Q K^T)*scale via WMMA  -> bf16 scores staged in LDS
//   pass2: vectorized LDS row scans: row max, then exp + row sum (2 lanes/row)
//   pass3: O = P V via WMMA reading A-fragments from LDS (ds_load_b128)
// 4 waves/block, 64KB LDS score slab per wave (256.5 KB of the 320KB WGP LDS)
// ---------------------------------------------------------------------------
#define ATTN_LDS_BYTES (4 * (16 * TSEQ * 2) + 4 * 128)

__global__ __launch_bounds__(128) void attention_kernel(
    const __bf16* __restrict__ Q,   // [B,QH,T,64]
    const __bf16* __restrict__ Kr,  // [B,KVH,T,64]
    const __bf16* __restrict__ Vt,  // [B,KVH,64,T]
    __bf16* __restrict__ O) {       // [B,T,1024]  (b, t, qh*64 + d)
  extern __shared__ char smem[];
  const int lane = threadIdx.x & 31;
  const int wave = threadIdx.x >> 5;
  __bf16* P = (__bf16*)(smem + (size_t)wave * (16 * TSEQ * 2));
  float* stats = (float*)(smem + 4 * (16 * TSEQ * 2) + wave * 128);
  // stats[0..15] = row sum

  const int id = blockIdx.x * 4 + wave;   // query-tile id
  const int ntile = TSEQ / 16;            // 128
  const int b = id / (QHEADS * ntile);
  const int rem = id % (QHEADS * ntile);
  const int qh = rem / ntile;
  const int tt = rem % ntile;
  const int kh = qh >> 2;                 // GROUP = 4
  const int t0 = tt * 16;

  const __bf16* Qb = Q + (((size_t)b * QHEADS + qh) * TSEQ + t0) * HDIM;
  const __bf16* Kb = Kr + (((size_t)b * KVHEADS + kh) * TSEQ) * HDIM;
  const __bf16* Vb = Vt + (((size_t)b * KVHEADS + kh) * HDIM) * TSEQ;

  const int half = lane >> 4;
  const int nlo = lane & 15;
  const float scale = 0.125f;             // 1/sqrt(64)

  // Q fragments (K-dim = 64 -> two 32-wide steps), reused for all key tiles
  const v16bf aq0 = load_a_frag(Qb, HDIM, 0, 0, lane);
  const v16bf aq1 = load_a_frag(Qb, HDIM, 0, 32, lane);

  // ---- pass 1: scaled scores into LDS ----
  for (int n0 = 0; n0 < TSEQ; n0 += 16) {
    v8f s = {};
    s = wmma_bf16(aq0, load_b_frag(Kb, HDIM, n0, 0, lane), s);
    s = wmma_bf16(aq1, load_b_frag(Kb, HDIM, n0, 32, lane), s);
    #pragma unroll
    for (int r = 0; r < 8; ++r) {
      P[(half * 8 + r) * TSEQ + n0 + nlo] = (__bf16)(s[r] * scale);
    }
  }

  // ---- pass 2: row max, then exp + row sum (2 lanes per row, 8-wide) ----
  {
    const int row = lane >> 1;
    const int c0 = (lane & 1) * (TSEQ / 2);
    __bf16* Prow = P + row * TSEQ + c0;

    float mx = -3.0e38f;
    for (int c = 0; c < TSEQ / 2; c += 8) {
      v8bf u = *reinterpret_cast<const v8bf*>(Prow + c);
      #pragma unroll
      for (int e = 0; e < 8; ++e) mx = fmaxf(mx, (float)u[e]);
    }
    mx = fmaxf(mx, __shfl_xor(mx, 1, 32));   // combine the two half-rows

    float sum = 0.f;
    for (int c = 0; c < TSEQ / 2; c += 8) {
      v8bf u = *reinterpret_cast<v8bf*>(Prow + c);
      #pragma unroll
      for (int e = 0; e < 8; ++e) {
        const float pv = __expf((float)u[e] - mx);
        sum += pv;
        u[e] = (__bf16)pv;
      }
      *reinterpret_cast<v8bf*>(Prow + c) = u;
    }
    sum += __shfl_xor(sum, 1, 32);
    if ((lane & 1) == 0) stats[row] = sum;
  }

  // ---- pass 3: O = P @ V (A from LDS, B contiguous from Vt) ----
  v8f acc[4] = {};
  for (int k0 = 0; k0 < TSEQ; k0 += 32) {
    const v16bf ap = load_a_frag(P, TSEQ, 0, k0, lane);   // ds_load_b128 x2
    #pragma unroll
    for (int j = 0; j < 4; ++j) {
      acc[j] = wmma_bf16(ap, load_b_frag(Vb, TSEQ, j * 16, k0, lane), acc[j]);
    }
  }

  float inv[8];
  #pragma unroll
  for (int r = 0; r < 8; ++r) inv[r] = 1.0f / stats[half * 8 + r];

  __bf16* Ob = O + ((size_t)b * TSEQ + t0) * DM + qh * HDIM;
  #pragma unroll
  for (int j = 0; j < 4; ++j) {
    #pragma unroll
    for (int r = 0; r < 8; ++r) {
      const int m = half * 8 + r;       // query row within tile
      const int d = j * 16 + nlo;
      Ob[(size_t)m * DM + d] = (__bf16)(acc[j][r] * inv[r]);
    }
  }
}

// ---------------------------------------------------------------------------
// Host launcher
// ---------------------------------------------------------------------------
static inline char* ws_bump(char*& p, size_t bytes) {
  char* r = p;
  p += (bytes + 255) & ~(size_t)255;
  return r;
}

extern "C" void kernel_launch(void* const* d_in, const int* in_sizes, int n_in,
                              void* d_out, int out_size, void* d_ws, size_t ws_size,
                              hipStream_t stream) {
  (void)in_sizes; (void)n_in; (void)out_size; (void)ws_size;

  const float* x     = (const float*)d_in[0];
  const float* cosT  = (const float*)d_in[1];
  const float* sinT  = (const float*)d_in[2];
  const float* Wq    = (const float*)d_in[3];
  const float* Wkv   = (const float*)d_in[4];
  const float* Wo    = (const float*)d_in[5];
  const float* ln1_g = (const float*)d_in[6];
  const float* ln1_b = (const float*)d_in[7];
  const float* ln2_g = (const float*)d_in[8];
  const float* ln2_b = (const float*)d_in[9];
  const float* W1    = (const float*)d_in[10];
  const float* b1    = (const float*)d_in[11];
  const float* W2    = (const float*)d_in[12];
  const float* b2    = (const float*)d_in[13];
  float* out = (float*)d_out;

  const int M = BATCH * TSEQ;                     // 8192 rows

  // workspace layout
  char* p = (char*)d_ws;
  __bf16* wq_bf  = (__bf16*)ws_bump(p, (size_t)DM * DM * 2);
  __bf16* wkv_bf = (__bf16*)ws_bump(p, (size_t)(2 * KVHEADS * HDIM) * DM * 2);
  __bf16* wo_bf  = (__bf16*)ws_bump(p, (size_t)DM * DM * 2);
  __bf16* w1_bf  = (__bf16*)ws_bump(p, (size_t)DFF * DM * 2);
  __bf16* w2_bf  = (__bf16*)ws_bump(p, (size_t)DM * DFF * 2);
  __bf16* h_bf   = (__bf16*)ws_bump(p, (size_t)M * DM * 2);
  float*  qlin   = (float*) ws_bump(p, (size_t)M * DM * 4);
  float*  kvlin  = (float*) ws_bump(p, (size_t)M * (2 * KVHEADS * HDIM) * 4);
  __bf16* Qr     = (__bf16*)ws_bump(p, (size_t)BATCH * QHEADS * TSEQ * HDIM * 2);
  __bf16* Kr     = (__bf16*)ws_bump(p, (size_t)BATCH * KVHEADS * TSEQ * HDIM * 2);
  __bf16* Vt     = (__bf16*)ws_bump(p, (size_t)BATCH * KVHEADS * HDIM * TSEQ * 2);
  __bf16* Obf    = (__bf16*)ws_bump(p, (size_t)M * DM * 2);
  float*  x2     = (float*) ws_bump(p, (size_t)M * DM * 4);
  __bf16* h2_bf  = (__bf16*)ws_bump(p, (size_t)M * DM * 2);
  __bf16* g_bf   = (__bf16*)ws_bump(p, (size_t)M * DFF * 2);

  // 1) weights -> bf16
  cvt_f32_bf16_kernel<<<(DM * DM + 255) / 256, 256, 0, stream>>>(Wq, wq_bf, DM * DM);
  cvt_f32_bf16_kernel<<<(512 * DM + 255) / 256, 256, 0, stream>>>(Wkv, wkv_bf, 512 * DM);
  cvt_f32_bf16_kernel<<<(DM * DM + 255) / 256, 256, 0, stream>>>(Wo, wo_bf, DM * DM);
  cvt_f32_bf16_kernel<<<(DFF * DM + 255) / 256, 256, 0, stream>>>(W1, w1_bf, DFF * DM);
  cvt_f32_bf16_kernel<<<(DM * DFF + 255) / 256, 256, 0, stream>>>(W2, w2_bf, DM * DFF);

  // 2) LN1 -> h (bf16)
  layernorm_kernel<<<M, 256, 0, stream>>>(x, ln1_g, ln1_b, h_bf);

  // 3) q = h Wq^T, kv = h Wkv^T (fp32)
  gemm_bf16_kernel<EPI_F32><<<dim3(DM / 128, M / 128), 128, 0, stream>>>(
      h_bf, wq_bf, nullptr, nullptr, qlin, nullptr, M, DM, DM);
  gemm_bf16_kernel<EPI_F32><<<dim3(512 / 128, M / 128), 128, 0, stream>>>(
      h_bf, wkv_bf, nullptr, nullptr, kvlin, nullptr, M, 512, DM);

  // 4) RoPE + head-major bf16 tensors
  rope_q_kernel<<<(BATCH * QHEADS * TSEQ * 32 + 255) / 256, 256, 0, stream>>>(
      qlin, cosT, sinT, Qr);
  rope_k_kernel<<<(BATCH * KVHEADS * TSEQ * 32 + 255) / 256, 256, 0, stream>>>(
      kvlin, cosT, sinT, Kr);
  v_transpose_kernel<<<(BATCH * KVHEADS * HDIM * TSEQ + 255) / 256, 256, 0, stream>>>(
      kvlin, Vt);

  // 5) attention (4 waves per block, one 16-query tile per wave)
  {
    const int qtiles = BATCH * QHEADS * (TSEQ / 16);   // 8192
    hipFuncSetAttribute((const void*)attention_kernel,
                        hipFuncAttributeMaxDynamicSharedMemorySize,
                        ATTN_LDS_BYTES);
    attention_kernel<<<qtiles / 4, 128, ATTN_LDS_BYTES, stream>>>(Qr, Kr, Vt, Obf);
  }

  // 6) x2 = x + O Wo^T
  gemm_bf16_kernel<EPI_RES_F32><<<dim3(DM / 128, M / 128), 128, 0, stream>>>(
      Obf, wo_bf, nullptr, x, x2, nullptr, M, DM, DM);

  // 7) LN2 -> h2 (bf16)
  layernorm_kernel<<<M, 256, 0, stream>>>(x2, ln2_g, ln2_b, h2_bf);

  // 8) g = gelu(h2 W1^T + b1) (bf16)
  gemm_bf16_kernel<EPI_GELU_BF16><<<dim3(DFF / 128, M / 128), 128, 0, stream>>>(
      h2_bf, w1_bf, b1, nullptr, nullptr, g_bf, M, DFF, DM);

  // 9) out = x2 + g W2^T + b2 (fp32)
  gemm_bf16_kernel<EPI_RES_F32><<<dim3(DM / 128, M / 128), 128, 0, stream>>>(
      g_bf, w2_bf, b2, x2, out, nullptr, M, DM, DFF);
}
